// LlamaEmbeddingModel_35270271435322
// MI455X (gfx1250) — compile-verified, matrix operand order checked by reading, not run on hardware
//
#include <hip/hip_runtime.h>
#include <hip/hip_bf16.h>
#include <math.h>

// ---------------- model constants ----------------
#define LAYERS 2
#define VOCAB  32000
#define DMODEL 2048
#define NHEAD  16
#define NKVH   8
#define HDIM   128
#define FFDIM  8192
#define LRANK  64
#define LSCALE 2.0f
#define POUT   768
#define BB     8
#define SS     512
#define MTOK   (BB * SS)      // 4096 tokens
#define THETA  500000.0f
#define RMSEPS 1e-5f

typedef __attribute__((ext_vector_type(16))) __bf16 bf16x16;
typedef __attribute__((ext_vector_type(8)))  float  f32x8;

// pack 4 floats -> 4 bf16 and store as one 8-byte LDS write
__device__ __forceinline__ void store_bf16x4(__bf16* dst, float4 f)
{
    union { __bf16 h[4]; uint2 u; } tmp;
    tmp.h[0] = (__bf16)f.x;
    tmp.h[1] = (__bf16)f.y;
    tmp.h[2] = (__bf16)f.z;
    tmp.h[3] = (__bf16)f.w;
    *(uint2*)dst = tmp.u;
}

// =====================================================================
// GEMM:  C[M,N] (=|+=) alpha * A[M,K] @ W[N,K]^T      (bf16 WMMA, f32 acc)
// block tile 64x64, 256 threads (8 waves): waveM in {0,1} x waveN in {0..3}
// each wave: 2 accumulators of 16x16 (32 rows x 16 cols)
// Staging: float4 (b128) global loads -> packed bf16 b64 DS stores.
// =====================================================================
#define TM 64
#define TN 64
#define TK 32

template <int ACCUM>
__global__ __launch_bounds__(256)
void gemm_wmma_kernel(const float* __restrict__ A, const float* __restrict__ W,
                      float* __restrict__ C, int N, int K, float alpha)
{
    __shared__ __bf16 As[TM][TK + 8];
    __shared__ __bf16 Bs[TN][TK + 8];

    const int tid   = threadIdx.x;
    const int lane  = tid & 31;
    const int wave  = tid >> 5;      // 0..7
    const int waveM = wave >> 2;     // 0..1
    const int waveN = wave & 3;      // 0..3
    const int half  = lane >> 4;     // 0/1
    const int l16   = lane & 15;

    const int blockM = blockIdx.y * TM;
    const int blockN = blockIdx.x * TN;

    f32x8 acc0 = {}; f32x8 acc1 = {};

    for (int k0 = 0; k0 < K; k0 += TK) {
        // cooperative stage: 64x32 of A and of W as bf16 (float4 quads)
#pragma unroll
        for (int g = 0; g < 2; ++g) {
            int q  = tid + g * 256;     // 0..511 quads
            int r  = q >> 3;            // 8 quads per 32-wide row
            int cq = (q & 7) << 2;
            float4 av = *(const float4*)&A[(size_t)(blockM + r) * K + k0 + cq];
            float4 wv = *(const float4*)&W[(size_t)(blockN + r) * K + k0 + cq];
            store_bf16x4(&As[r][cq], av);
            store_bf16x4(&Bs[r][cq], wv);
        }
        // prefetch next K-tile while this one is consumed
        if (k0 + TK < K) {
            int r  = tid >> 3;
            int cq = (tid & 7) << 2;
            __builtin_prefetch(&A[(size_t)(blockM + r) * K + k0 + TK + cq], 0, 1);
            __builtin_prefetch(&W[(size_t)(blockN + r) * K + k0 + TK + cq], 0, 1);
        }
        __syncthreads();

        bf16x16 afrag0, afrag1, bfrag;
        const int arow0 = waveM * 32 + l16;
        const int arow1 = arow0 + 16;
        const int brow  = waveN * 16 + l16;
#pragma unroll
        for (int v = 0; v < 8; ++v) {
            int kk = ((v & 4) ? 16 : 0) + half * 8 + 2 * (v & 3);
            afrag0[2*v]   = As[arow0][kk];
            afrag0[2*v+1] = As[arow0][kk+1];
            afrag1[2*v]   = As[arow1][kk];
            afrag1[2*v+1] = As[arow1][kk+1];
            bfrag[2*v]    = Bs[brow][kk];
            bfrag[2*v+1]  = Bs[brow][kk+1];
        }
        acc0 = __builtin_amdgcn_wmma_f32_16x16x32_bf16(false, afrag0, false, bfrag,
                                                       (short)0, acc0, false, false);
        acc1 = __builtin_amdgcn_wmma_f32_16x16x32_bf16(false, afrag1, false, bfrag,
                                                       (short)0, acc1, false, false);
        __syncthreads();
    }

    const int col = blockN + waveN * 16 + l16;
#pragma unroll
    for (int r = 0; r < 8; ++r) {
        int row0 = blockM + waveM * 32 + r + half * 8;
        int row1 = row0 + 16;
        size_t i0 = (size_t)row0 * N + col;
        size_t i1 = (size_t)row1 * N + col;
        float v0 = alpha * acc0[r];
        float v1 = alpha * acc1[r];
        if (ACCUM) {
            C[i0] += v0;
            C[i1] += v1;
        } else {
            C[i0] = v0;
            C[i1] = v1;
        }
    }
}

// =====================================================================
// Embedding gather: h[t,:] = tok_emb[ids[t],:]
// =====================================================================
__global__ void embed_kernel(const int* __restrict__ ids,
                             const float* __restrict__ emb,
                             float* __restrict__ h)
{
    int t = blockIdx.x;
    int id = ids[t];
    const float4* src = (const float4*)(emb + (size_t)id * DMODEL);
    float4* dst = (float4*)(h + (size_t)t * DMODEL);
    for (int i = threadIdx.x; i < DMODEL / 4; i += blockDim.x) dst[i] = src[i];
}

// =====================================================================
// RMSNorm per row (one block/row, 256 threads)
// =====================================================================
__global__ void rmsnorm_kernel(const float* __restrict__ X,
                               const float* __restrict__ w,
                               float* __restrict__ Y, int Dd)
{
    __shared__ float red[256];
    int row = blockIdx.x;
    const float* x = X + (size_t)row * Dd;
    float s = 0.f;
    for (int i = threadIdx.x; i < Dd; i += blockDim.x) { float v = x[i]; s += v * v; }
    red[threadIdx.x] = s;
    __syncthreads();
    for (int off = 128; off > 0; off >>= 1) {
        if (threadIdx.x < off) red[threadIdx.x] += red[threadIdx.x + off];
        __syncthreads();
    }
    float inv = rsqrtf(red[0] / (float)Dd + RMSEPS);
    for (int i = threadIdx.x; i < Dd; i += blockDim.x)
        Y[(size_t)row * Dd + i] = x[i] * inv * w[i];
}

// =====================================================================
// RoPE in place: x is [MTOK, heads*HDIM]
// =====================================================================
__global__ void rope_kernel(float* __restrict__ x, int heads)
{
    int m = blockIdx.x;
    float pos = (float)(m % SS);
    for (int i = threadIdx.x; i < heads * (HDIM / 2); i += blockDim.x) {
        int hh = i / (HDIM / 2);
        int d  = i % (HDIM / 2);
        float inv = powf(THETA, -2.0f * (float)d / (float)HDIM);
        float ang = pos * inv;
        float c = cosf(ang), sn = sinf(ang);
        size_t base = (size_t)m * heads * HDIM + (size_t)hh * HDIM;
        float x1 = x[base + d];
        float x2 = x[base + d + HDIM / 2];
        x[base + d]            = x1 * c - x2 * sn;
        x[base + d + HDIM / 2] = x2 * c + x1 * sn;
    }
}

// =====================================================================
// Flash-style causal attention with WMMA for QK^T and P@V.
// grid = (B*H, S/64), block = 128 (4 waves); each wave owns 16 queries.
// Key tiles of 32 staged in LDS as bf16 (shared K/V across waves).
// =====================================================================
__global__ __launch_bounds__(128)
void attn_kernel(const float* __restrict__ Q, const float* __restrict__ Kb,
                 const float* __restrict__ Vb, const int* __restrict__ amask,
                 float* __restrict__ O)
{
    __shared__ __bf16 Ks[32][HDIM + 8];
    __shared__ __bf16 Vs[32][HDIM + 8];
    __shared__ __bf16 Ps[4][16][36];   // per-wave P scratch (16 q x 32 k)

    const int tid  = threadIdx.x;
    const int lane = tid & 31;
    const int wave = tid >> 5;
    const int half = lane >> 4;
    const int l16  = lane & 15;

    const int bh  = blockIdx.x;
    const int b   = bh / NHEAD;
    const int hh  = bh % NHEAD;
    const int kvh = hh / (NHEAD / NKVH);
    const int q0  = blockIdx.y * 64;
    const int qw  = q0 + wave * 16;

    // load Q fragments (16x128 -> 4 A-fragments of 16x32)
    bf16x16 qfrag[4];
    {
        const float* qrow = Q + (size_t)(b * SS + qw + l16) * (NHEAD * HDIM)
                              + (size_t)hh * HDIM;
#pragma unroll
        for (int f = 0; f < 4; ++f)
#pragma unroll
            for (int v = 0; v < 8; ++v) {
                int kk = f * 32 + ((v & 4) ? 16 : 0) + half * 8 + 2 * (v & 3);
                qfrag[f][2*v]   = (__bf16)qrow[kk];
                qfrag[f][2*v+1] = (__bf16)qrow[kk+1];
            }
    }

    float rmax[8], rsum[8];
#pragma unroll
    for (int r = 0; r < 8; ++r) { rmax[r] = -1e30f; rsum[r] = 0.f; }
    f32x8 oacc[8];
#pragma unroll
    for (int f = 0; f < 8; ++f) oacc[f] = (f32x8){};

    const float sscale = 0.08838834764831845f;  // 1/sqrt(128)
    const int ntiles = (q0 + 64) / 32;          // causal coverage for this block

    for (int t = 0; t < ntiles; ++t) {
        const int kbase = t * 32;
        // stage K,V tiles (32 x 128) via float4 quads -> packed bf16 b64 DS stores
#pragma unroll
        for (int j = 0; j < 8; ++j) {
            int q  = tid + j * 128;     // 0..1023 quads
            int kr = q >> 5;            // 32 quads per 128-wide row
            int cq = (q & 31) << 2;
            size_t gi = (size_t)(b * SS + kbase + kr) * (NKVH * HDIM)
                      + (size_t)kvh * HDIM + cq;
            store_bf16x4(&Ks[kr][cq], *(const float4*)&Kb[gi]);
            store_bf16x4(&Vs[kr][cq], *(const float4*)&Vb[gi]);
        }
        __syncthreads();

        // scores 16x32 -> two 16x16 f32 accumulators
        f32x8 sacc[2]; sacc[0] = (f32x8){}; sacc[1] = (f32x8){};
#pragma unroll
        for (int nt = 0; nt < 2; ++nt) {
            int krow = nt * 16 + l16;
#pragma unroll
            for (int f = 0; f < 4; ++f) {
                bf16x16 kfrag;
#pragma unroll
                for (int v = 0; v < 8; ++v) {
                    int kk = f * 32 + ((v & 4) ? 16 : 0) + half * 8 + 2 * (v & 3);
                    kfrag[2*v]   = Ks[krow][kk];
                    kfrag[2*v+1] = Ks[krow][kk+1];
                }
                sacc[nt] = __builtin_amdgcn_wmma_f32_16x16x32_bf16(
                    false, qfrag[f], false, kfrag, (short)0, sacc[nt], false, false);
            }
        }

        // mask + tile row-max
        float tmax[8], sval[2][8];
#pragma unroll
        for (int r = 0; r < 8; ++r) tmax[r] = -1e30f;
#pragma unroll
        for (int nt = 0; nt < 2; ++nt) {
            int kidx = kbase + nt * 16 + l16;
            int ok = (amask[b * SS + kidx] > 0);
#pragma unroll
            for (int r = 0; r < 8; ++r) {
                int qidx = qw + r + half * 8;
                float sv = sacc[nt][r] * sscale;
                if (kidx > qidx || !ok) sv = -1e30f;
                sval[nt][r] = sv;
                tmax[r] = fmaxf(tmax[r], sv);
            }
        }
#pragma unroll
        for (int mm = 1; mm <= 8; mm <<= 1)
#pragma unroll
            for (int r = 0; r < 8; ++r)
                tmax[r] = fmaxf(tmax[r], __shfl_xor(tmax[r], mm, 32));

        float fac[8], tsum[8];
#pragma unroll
        for (int r = 0; r < 8; ++r) {
            float nm = fmaxf(rmax[r], tmax[r]);
            fac[r] = expf(rmax[r] - nm);
            rmax[r] = nm;
            tsum[r] = 0.f;
        }
        // exp -> LDS (bf16), per-wave private region: DS in-order within wave
#pragma unroll
        for (int nt = 0; nt < 2; ++nt)
#pragma unroll
            for (int r = 0; r < 8; ++r) {
                float p = expf(sval[nt][r] - rmax[r]);
                tsum[r] += p;
                Ps[wave][r + half * 8][nt * 16 + l16] = (__bf16)p;
            }
#pragma unroll
        for (int mm = 1; mm <= 8; mm <<= 1)
#pragma unroll
            for (int r = 0; r < 8; ++r)
                tsum[r] += __shfl_xor(tsum[r], mm, 32);
#pragma unroll
        for (int r = 0; r < 8; ++r) rsum[r] = rsum[r] * fac[r] + tsum[r];
#pragma unroll
        for (int f = 0; f < 8; ++f)
#pragma unroll
            for (int r = 0; r < 8; ++r)
                oacc[f][r] *= fac[r];

        // P fragment (A-matrix 16x32)
        bf16x16 pfrag;
#pragma unroll
        for (int v = 0; v < 8; ++v) {
            int kk = ((v & 4) ? 16 : 0) + half * 8 + 2 * (v & 3);
            pfrag[2*v]   = Ps[wave][l16][kk];
            pfrag[2*v+1] = Ps[wave][l16][kk+1];
        }
        // P @ V : 8 column tiles of 16
#pragma unroll
        for (int f = 0; f < 8; ++f) {
            bf16x16 vfrag;
#pragma unroll
            for (int v = 0; v < 8; ++v) {
                int kk = ((v & 4) ? 16 : 0) + half * 8 + 2 * (v & 3);
                vfrag[2*v]   = Vs[kk][f * 16 + l16];
                vfrag[2*v+1] = Vs[kk + 1][f * 16 + l16];
            }
            oacc[f] = __builtin_amdgcn_wmma_f32_16x16x32_bf16(
                false, pfrag, false, vfrag, (short)0, oacc[f], false, false);
        }
        __syncthreads();
    }

    // write O = acc / rowsum into [MTOK, H*HD] layout
#pragma unroll
    for (int f = 0; f < 8; ++f)
#pragma unroll
        for (int r = 0; r < 8; ++r) {
            int qidx = qw + r + half * 8;
            size_t oi = (size_t)(b * SS + qidx) * (NHEAD * HDIM)
                      + (size_t)hh * HDIM + f * 16 + l16;
            O[oi] = oacc[f][r] / rsum[r];
        }
}

// =====================================================================
// g = silu(g) * u
// =====================================================================
__global__ void silu_mul_kernel(float* __restrict__ g, const float* __restrict__ u,
                                size_t n)
{
    size_t i = (size_t)blockIdx.x * blockDim.x + threadIdx.x;
    if (i < n) {
        float x = g[i];
        float s = x / (1.f + expf(-x));
        g[i] = s * u[i];
    }
}

// =====================================================================
// masked mean pool over sequence: pooled[b,d]
// =====================================================================
__global__ void pool_kernel(const float* __restrict__ xn,
                            const int* __restrict__ amask,
                            float* __restrict__ pooled)
{
    int b = blockIdx.x;
    float cnt = 0.f;
    for (int s = 0; s < SS; ++s) cnt += (amask[b * SS + s] > 0) ? 1.f : 0.f;
    cnt = fmaxf(cnt, 1e-9f);
    for (int d = threadIdx.x; d < DMODEL; d += blockDim.x) {
        float acc = 0.f;
        for (int s = 0; s < SS; ++s)
            if (amask[b * SS + s] > 0)
                acc += xn[(size_t)(b * SS + s) * DMODEL + d];
        pooled[(size_t)b * DMODEL + d] = acc / cnt;
    }
}

// =====================================================================
// out[b,p] = pooled[b,:] . projW[p,:] + projb[p]
// =====================================================================
__global__ void proj_kernel(const float* __restrict__ pooled,
                            const float* __restrict__ Wp,
                            const float* __restrict__ bp,
                            float* __restrict__ out)
{
    int idx = blockIdx.x * blockDim.x + threadIdx.x;
    if (idx >= BB * POUT) return;
    int b = idx / POUT, p = idx % POUT;
    float acc = bp[p];
    const float* x = pooled + (size_t)b * DMODEL;
    const float* w = Wp + (size_t)p * DMODEL;
    for (int d = 0; d < DMODEL; ++d) acc += x[d] * w[d];
    out[idx] = acc;
}

// =====================================================================
// L2 normalize rows of out[B, POUT] in place
// =====================================================================
__global__ void l2norm_kernel(float* __restrict__ out)
{
    __shared__ float red[256];
    int b = blockIdx.x;
    float s = 0.f;
    for (int p = threadIdx.x; p < POUT; p += blockDim.x) {
        float v = out[(size_t)b * POUT + p];
        s += v * v;
    }
    red[threadIdx.x] = s;
    __syncthreads();
    for (int off = 128; off > 0; off >>= 1) {
        if (threadIdx.x < off) red[threadIdx.x] += red[threadIdx.x + off];
        __syncthreads();
    }
    float inv = 1.f / fmaxf(sqrtf(red[0]), 1e-12f);
    for (int p = threadIdx.x; p < POUT; p += blockDim.x)
        out[(size_t)b * POUT + p] *= inv;
}

// =====================================================================
// host driver
// =====================================================================
extern "C" void kernel_launch(void* const* d_in, const int* in_sizes, int n_in,
                              void* d_out, int out_size, void* d_ws, size_t ws_size,
                              hipStream_t stream)
{
    (void)in_sizes; (void)n_in; (void)out_size; (void)ws_size;

    const int*   ids   = (const int*)  d_in[0];
    const int*   amask = (const int*)  d_in[1];
    const float* tok   = (const float*)d_in[2];
    const float* Wq = (const float*)d_in[3];
    const float* Wk = (const float*)d_in[4];
    const float* Wv = (const float*)d_in[5];
    const float* Wo = (const float*)d_in[6];
    const float* Wg = (const float*)d_in[7];
    const float* Wu = (const float*)d_in[8];
    const float* Wd = (const float*)d_in[9];
    const float* Aq = (const float*)d_in[10];
    const float* Bq = (const float*)d_in[11];
    const float* Ak = (const float*)d_in[12];
    const float* Bk = (const float*)d_in[13];
    const float* Av = (const float*)d_in[14];
    const float* Bv = (const float*)d_in[15];
    const float* Ao = (const float*)d_in[16];
    const float* Bo = (const float*)d_in[17];
    const float* Ag = (const float*)d_in[18];
    const float* Bg = (const float*)d_in[19];
    const float* Au = (const float*)d_in[20];
    const float* Bu = (const float*)d_in[21];
    const float* Ad = (const float*)d_in[22];
    const float* Bd = (const float*)d_in[23];
    const float* ln1 = (const float*)d_in[24];
    const float* ln2 = (const float*)d_in[25];
    const float* lnf = (const float*)d_in[26];
    const float* projW = (const float*)d_in[27];
    const float* projb = (const float*)d_in[28];
    float* out = (float*)d_out;

    // workspace carve-up (floats)
    float* ws = (float*)d_ws;
    float* h      = ws;                                   // MTOK*DMODEL
    float* xn     = h   + (size_t)MTOK * DMODEL;
    float* qb     = xn  + (size_t)MTOK * DMODEL;          // MTOK*2048
    float* kb     = qb  + (size_t)MTOK * (NHEAD * HDIM);  // MTOK*1024
    float* vb     = kb  + (size_t)MTOK * (NKVH * HDIM);
    float* ob     = vb  + (size_t)MTOK * (NKVH * HDIM);   // MTOK*2048
    float* gb     = ob  + (size_t)MTOK * (NHEAD * HDIM);  // MTOK*FF
    float* ub     = gb  + (size_t)MTOK * FFDIM;
    float* lt     = ub  + (size_t)MTOK * FFDIM;           // MTOK*R
    float* pooled = lt  + (size_t)MTOK * LRANK;           // B*D

    auto gemm = [&](const float* A, const float* W, float* C,
                    int M, int N, int K, float alpha, int accum) {
        dim3 g(N / TN, M / TM);
        if (accum) gemm_wmma_kernel<1><<<g, 256, 0, stream>>>(A, W, C, N, K, alpha);
        else       gemm_wmma_kernel<0><<<g, 256, 0, stream>>>(A, W, C, N, K, alpha);
    };

    embed_kernel<<<MTOK, 256, 0, stream>>>(ids, tok, h);

    for (int l = 0; l < LAYERS; ++l) {
        const size_t oDD  = (size_t)l * DMODEL * DMODEL;          // 2048x2048
        const size_t oKV  = (size_t)l * (NKVH * HDIM) * DMODEL;   // 1024x2048
        const size_t oFF  = (size_t)l * FFDIM * DMODEL;           // 8192x2048
        const size_t oRD  = (size_t)l * LRANK * DMODEL;           // 64x2048
        const size_t oRF  = (size_t)l * LRANK * FFDIM;            // 64x8192
        const size_t oDR  = (size_t)l * DMODEL * LRANK;           // 2048x64
        const size_t oKVR = (size_t)l * (NKVH * HDIM) * LRANK;    // 1024x64
        const size_t oFR  = (size_t)l * FFDIM * LRANK;            // 8192x64

        // ---- attention block ----
        rmsnorm_kernel<<<MTOK, 256, 0, stream>>>(h, ln1 + (size_t)l * DMODEL, xn, DMODEL);

        gemm(xn, Wq + oDD, qb, MTOK, NHEAD * HDIM, DMODEL, 1.f, 0);
        gemm(xn, Aq + oRD, lt, MTOK, LRANK, DMODEL, 1.f, 0);
        gemm(lt, Bq + oDR, qb, MTOK, NHEAD * HDIM, LRANK, LSCALE, 1);

        gemm(xn, Wk + oKV, kb, MTOK, NKVH * HDIM, DMODEL, 1.f, 0);
        gemm(xn, Ak + oRD, lt, MTOK, LRANK, DMODEL, 1.f, 0);
        gemm(lt, Bk + oKVR, kb, MTOK, NKVH * HDIM, LRANK, LSCALE, 1);

        gemm(xn, Wv + oKV, vb, MTOK, NKVH * HDIM, DMODEL, 1.f, 0);
        gemm(xn, Av + oRD, lt, MTOK, LRANK, DMODEL, 1.f, 0);
        gemm(lt, Bv + oKVR, vb, MTOK, NKVH * HDIM, LRANK, LSCALE, 1);

        rope_kernel<<<MTOK, 128, 0, stream>>>(qb, NHEAD);
        rope_kernel<<<MTOK, 128, 0, stream>>>(kb, NKVH);

        {
            dim3 g(BB * NHEAD, SS / 64);
            attn_kernel<<<g, 128, 0, stream>>>(qb, kb, vb, amask, ob);
        }

        gemm(ob, Wo + oDD, h, MTOK, DMODEL, NHEAD * HDIM, 1.f, 1);
        gemm(ob, Ao + oRD, lt, MTOK, LRANK, NHEAD * HDIM, 1.f, 0);
        gemm(lt, Bo + oDR, h, MTOK, DMODEL, LRANK, LSCALE, 1);

        // ---- MLP block ----
        rmsnorm_kernel<<<MTOK, 256, 0, stream>>>(h, ln2 + (size_t)l * DMODEL, xn, DMODEL);

        gemm(xn, Wg + oFF, gb, MTOK, FFDIM, DMODEL, 1.f, 0);
        gemm(xn, Ag + oRD, lt, MTOK, LRANK, DMODEL, 1.f, 0);
        gemm(lt, Bg + oFR, gb, MTOK, FFDIM, LRANK, LSCALE, 1);

        gemm(xn, Wu + oFF, ub, MTOK, FFDIM, DMODEL, 1.f, 0);
        gemm(xn, Au + oRD, lt, MTOK, LRANK, DMODEL, 1.f, 0);
        gemm(lt, Bu + oFR, ub, MTOK, FFDIM, LRANK, LSCALE, 1);

        {
            size_t n = (size_t)MTOK * FFDIM;
            silu_mul_kernel<<<(int)((n + 255) / 256), 256, 0, stream>>>(gb, ub, n);
        }

        gemm(gb, Wd + (size_t)l * DMODEL * FFDIM, h, MTOK, DMODEL, FFDIM, 1.f, 1);
        gemm(gb, Ad + oRF, lt, MTOK, LRANK, FFDIM, 1.f, 0);
        gemm(lt, Bd + oDR, h, MTOK, DMODEL, LRANK, LSCALE, 1);
    }

    // ---- head ----
    rmsnorm_kernel<<<MTOK, 256, 0, stream>>>(h, lnf, xn, DMODEL);
    pool_kernel<<<BB, 256, 0, stream>>>(xn, amask, pooled);
    proj_kernel<<<(BB * POUT + 255) / 256, 256, 0, stream>>>(pooled, projW, projb, out);
    l2norm_kernel<<<BB, 256, 0, stream>>>(out);
}
